// MatrixFactorization_54984171323490
// MI455X (gfx1250) — matrix operand order repaired and checked by reading, not compiled, not run on hardware
//
#include <hip/hip_runtime.h>

// ---------------- types ----------------
typedef __bf16 bf16_t;
typedef __attribute__((ext_vector_type(16))) __bf16 v16bf;
typedef __attribute__((ext_vector_type(8)))  float  v8f;

// ---------------- problem constants ----------------
#define NUM_USERS 10000
#define NUM_ITEMS 20000
#define DIM       64
#define CLUSTER   256
#define BATCH     8192
#define ROWLEN    (NUM_ITEMS + 1)   // 20001 label columns
#define KTILES    (NUM_ITEMS / 32)  // 625
#define NTILES    (DIM / 16)        // 4
#define MTILES    (BATCH / 16)      // 512
#define FRAG_ELEMS ((long)KTILES * NTILES * 32 * 16)  // 1,280,000 bf16 per array

// =====================================================================
// Kernel 1: repack item_table (f32) into pre-swizzled WMMA B-fragments.
// B is 32x16 (KxN) bf16 per tile. ISA layout: lane L -> N = L&15,
// element e -> K = 8*half + (e&7) + 16*(e>>3), half = L>>4 (same order as A).
// Split each f32 into hi/lo bf16 so hi+lo ~= f32 (16-bit mantissa).
// Fragment storage: [(kt*4+nt)*32 + lane] * 16 contiguous bf16 per lane.
// =====================================================================
__global__ void prep_b_frags(const float* __restrict__ item_table,
                             bf16_t* __restrict__ bhi,
                             bf16_t* __restrict__ blo) {
  int gid = blockIdx.x * blockDim.x + threadIdx.x;
  if (gid >= KTILES * NTILES * 32) return;
  int lane = gid & 31;
  int nt   = (gid >> 5) & 3;
  int kt   = gid >> 7;
  int half = lane >> 4;
  int col  = nt * 16 + (lane & 15);
  long base = (long)gid * 16;
#pragma unroll
  for (int e = 0; e < 16; ++e) {
    int k = kt * 32 + 8 * half + (e & 7) + 16 * (e >> 3);  // < 20000: item_table[:-1]
    float v  = item_table[(long)k * DIM + col];
    bf16_t h = (bf16_t)v;
    float  r = v - (float)h;
    bhi[base + e] = h;
    blo[base + e] = (bf16_t)r;
  }
}

// =====================================================================
// Kernel 2: deterministic segment-mean cluster centers.
// One block per cluster, 256 threads = 4 item-groups x 64 dims,
// fixed-order reduction (no float atomics -> bitwise deterministic).
// =====================================================================
__global__ void cluster_centers(const float* __restrict__ item_table,
                                const int* __restrict__ cluster_ids,
                                float* __restrict__ centers) {
  int c = blockIdx.x;          // 0..255
  int d = threadIdx.x & 63;    // dim
  int g = threadIdx.x >> 6;    // 0..3 item group
  float acc = 0.0f;
  int cnt = 0;
  for (int i = g; i < ROWLEN; i += 4) {
    if (cluster_ids[i] == c) {
      acc += item_table[(long)i * DIM + d];
      ++cnt;
    }
  }
  __shared__ float sacc[256];
  __shared__ int   scnt[4];
  sacc[threadIdx.x] = acc;
  if (d == 0) scnt[g] = cnt;
  __syncthreads();
  if (g == 0) {
    float tot = sacc[d] + sacc[64 + d] + sacc[128 + d] + sacc[192 + d];
    int   n   = scnt[0] + scnt[1] + scnt[2] + scnt[3];
    centers[c * DIM + d] = tot / fmaxf((float)n, 1.0f);
  }
}

// =====================================================================
// Kernel 3: the five gather outputs.
// =====================================================================
__global__ void gather_embs(const int* __restrict__ user,
                            const int* __restrict__ pos,
                            const int* __restrict__ neg,
                            const int* __restrict__ cluster_ids,
                            const float* __restrict__ user_table,
                            const float* __restrict__ item_table,
                            const float* __restrict__ centers,
                            float* __restrict__ out) {
  int gid = blockIdx.x * blockDim.x + threadIdx.x;
  if (gid >= BATCH * DIM) return;
  int b = gid >> 6;
  int d = gid & 63;
  const long S = (long)BATCH * DIM;
  int u = user[b], p = pos[b], n = neg[b];
  out[gid]         = user_table[(long)u * DIM + d];
  out[S + gid]     = item_table[(long)p * DIM + d];
  out[2 * S + gid] = item_table[(long)n * DIM + d];
  out[3 * S + gid] = centers[cluster_ids[p] * DIM + d];
  out[4 * S + gid] = centers[cluster_ids[n] * DIM + d];
}

// =====================================================================
// Kernel 4: uni_center = (train_label[user][:, :20000] @ item_table[:20000]) / num_rel
// M=8192, K=20000, N=64. One wave per 16-row M-tile, full N in 4 v8f accs.
// Labels {0,1} are exact in bf16; B uses hi/lo bf16 split -> 2 WMMAs/tile.
// A layout (16-bit A 16x32): lane L -> M=L&15; e<8 -> K=8*half+e,
// e>=8 -> K=16+8*half+(e-8): two contiguous 8-float runs per lane.
// Row base = user*20001 floats (odd stride) -> scalar f32 loads (alignment).
// num_rel accumulated from the same loads (exact: integer f32 < 2^24),
// reduced across halves with wave32 shfl_xor(16), plus last column 20000.
// =====================================================================
__global__ void __launch_bounds__(128, 1)
label_gemm(const int* __restrict__ user,
           const float* __restrict__ train_label,
           const bf16_t* __restrict__ bhi,
           const bf16_t* __restrict__ blo,
           float* __restrict__ out5) {
  const int lane  = threadIdx.x & 31;
  const int wave  = threadIdx.x >> 5;         // 0..3
  const int mtile = blockIdx.x * 4 + wave;    // 0..511
  const int half  = lane >> 4;
  const int m     = lane & 15;

  const float* __restrict__ arow =
      train_label + (long)user[mtile * 16 + m] * ROWLEN;

  v8f acc[NTILES] = {};
  float labelsum = 0.0f;

  for (int kt = 0; kt < KTILES; ++kt) {
    const int kbase = kt * 32;
    const float* p0 = arow + kbase + 8 * half;   // K = 8*half + 0..7
    const float* p1 = p0 + 16;                   // K = 16 + 8*half + 0..7

    // prefetch A ~8 k-tiles ahead (speculative, OOB dropped)
    __builtin_prefetch(p0 + 256, 0, 1);

    v16bf a;
#pragma unroll
    for (int e = 0; e < 8; ++e) {
      float v0 = p0[e];
      float v1 = p1[e];
      labelsum += v0 + v1;      // exact: labels are 0/1
      a[e]     = (bf16_t)v0;    // exact in bf16
      a[8 + e] = (bf16_t)v1;
    }

    const long fb = (long)kt * 2048 + (long)lane * 16;  // ((kt*4+nt)*32+lane)*16
#pragma unroll
    for (int nt = 0; nt < NTILES; ++nt) {
      const v16bf bh = *(const v16bf*)(bhi + fb + (long)nt * 512);
      const v16bf bl = *(const v16bf*)(blo + fb + (long)nt * 512);
      acc[nt] = __builtin_amdgcn_wmma_f32_16x16x32_bf16(
          false, a, false, bh, (short)0, acc[nt], false, false);
      acc[nt] = __builtin_amdgcn_wmma_f32_16x16x32_bf16(
          false, a, false, bl, (short)0, acc[nt], false, false);
    }
  }

  // num_rel per row: both halves + untouched last label column (k = 20000)
  float rowsum = labelsum + __shfl_xor(labelsum, 16, 32) + arow[NUM_ITEMS];
  // lane L now holds num_rel for row m = L & 15

  // C/D layout: VGPR r, lane L -> M = r + 8*half, N = L & 15
#pragma unroll
  for (int r = 0; r < 8; ++r) {
    int   mr   = r + 8 * half;
    float nrel = __shfl(rowsum, mr, 32);
    long  bg   = (long)(mtile * 16 + mr) * DIM;
#pragma unroll
    for (int nt = 0; nt < NTILES; ++nt) {
      out5[bg + nt * 16 + m] = acc[nt][r] / nrel;
    }
  }
}

// =====================================================================
// Launch
// =====================================================================
extern "C" void kernel_launch(void* const* d_in, const int* in_sizes, int n_in,
                              void* d_out, int out_size, void* d_ws, size_t ws_size,
                              hipStream_t stream) {
  (void)in_sizes; (void)n_in; (void)out_size; (void)ws_size;

  const int*   user        = (const int*)d_in[0];
  const int*   pos         = (const int*)d_in[1];
  const int*   neg         = (const int*)d_in[2];
  const int*   cluster_ids = (const int*)d_in[3];
  const float* user_table  = (const float*)d_in[4];
  const float* item_table  = (const float*)d_in[5];
  const float* train_label = (const float*)d_in[6];
  float*       out         = (float*)d_out;

  // workspace layout: [bhi frags 2.56MB][blo frags 2.56MB][centers 64KB]
  bf16_t* bhi     = (bf16_t*)d_ws;
  bf16_t* blo     = bhi + FRAG_ELEMS;
  float*  centers = (float*)((char*)d_ws + 2 * FRAG_ELEMS * sizeof(bf16_t));

  const int prep_threads = KTILES * NTILES * 32;  // 80000
  prep_b_frags<<<(prep_threads + 255) / 256, 256, 0, stream>>>(item_table, bhi, blo);
  cluster_centers<<<CLUSTER, 256, 0, stream>>>(item_table, cluster_ids, centers);
  gather_embs<<<(BATCH * DIM + 255) / 256, 256, 0, stream>>>(
      user, pos, neg, cluster_ids, user_table, item_table, centers, out);
  label_gemm<<<MTILES / 4, 128, 0, stream>>>(
      user, train_label, bhi, blo, out + 5ull * BATCH * DIM);
}
